// FusedMoE_85890755985610
// MI455X (gfx1250) — matrix operand order repaired and checked by reading, not compiled, not run on hardware
//
#include <hip/hip_runtime.h>
#include <hip/hip_bf16.h>

// Problem constants (fixed by the reference).
constexpr int T_ = 512;
constexpr int H_ = 2048;
constexpr int E_ = 8;
constexpr int I_ = 1024;

// Tiling.
constexpr int BM  = 64;   // rows (token slots) per block
constexpr int BN  = 32;   // output columns per block (gemm1: gate cols, +matching up cols)
constexpr int BK  = 64;   // K chunk per LDS stage
constexpr int LDA = 72;   // LDS row stride in halves (144B: 16B-aligned, staggers banks)
constexpr int LDB = 72;

typedef __attribute__((ext_vector_type(16))) __bf16 v16bf;
typedef __attribute__((ext_vector_type(8)))  float  v8f;

// float -> bf16 (round-to-nearest-even), scalar.
static __device__ __forceinline__ unsigned short f2bf(float f) {
    unsigned u = __float_as_uint(f);
    unsigned r = ((u >> 16) & 1u) + 0x7fffu;
    return (unsigned short)((u + r) >> 16);
}

// Pack two floats into two bf16 (RNE bias adds + one v_perm_b32 merge).
static __device__ __forceinline__ unsigned pack2_bf16(float x, float y) {
    unsigned ux = __float_as_uint(x);
    unsigned uy = __float_as_uint(y);
    ux += ((ux >> 16) & 1u) + 0x7fffu;
    uy += ((uy >> 16) & 1u) + 0x7fffu;
    // bytes: [0]=ux.b2 [1]=ux.b3 [2]=uy.b2 [3]=uy.b3  (sel 0-3 -> src1, 4-7 -> src0)
    return __builtin_amdgcn_perm(uy, ux, 0x07060302u);
}

// Convert a float4 to 4 bf16 and store 8 bytes (dst is 8B-aligned by layout).
static __device__ __forceinline__ void st_bf4(unsigned short* d, float4 v) {
    unsigned* p = (unsigned*)d;
    p[0] = pack2_bf16(v.x, v.y);
    p[1] = pack2_bf16(v.z, v.w);
}

// A fragment, 16x32 bf16 (M x K), per ISA 7.12.2:
// lane L: row m = L&15, half = L>>4; K = 8*half + {0..7} and 16 + 8*half + {0..7}.
static __device__ __forceinline__ v16bf load_frag_a(const unsigned short* base, int lane, int stride) {
    const int m = lane & 15, half = lane >> 4;
    union { v16bf v; uint4 q[2]; } u;
    u.q[0] = *(const uint4*)(base + m * stride + half * 8);
    u.q[1] = *(const uint4*)(base + m * stride + half * 8 + 16);
    return u.v;
}

// B fragment, 32x16 bf16 (K x N), B stored in LDS as [n][k] (k contiguous):
// lane L: col n = L&15, khalf = L>>4; K = 8*khalf + {0..7} and 16 + 8*khalf + {0..7}.
static __device__ __forceinline__ v16bf load_frag_b(const unsigned short* base, int lane, int stride) {
    const int n = lane & 15, kh = lane >> 4;
    union { v16bf v; uint4 q[2]; } u;
    u.q[0] = *(const uint4*)(base + n * stride + kh * 8);
    u.q[1] = *(const uint4*)(base + n * stride + kh * 8 + 16);
    return u.v;
}

static __device__ __forceinline__ v8f wmma_bf16(v16bf a, v16bf b, v8f c) {
    return __builtin_amdgcn_wmma_f32_16x16x32_bf16(false, a, false, b, (short)0, c, false, false);
}

// ---------------------------------------------------------------------------
// Step 1: routing. softmax over E=8, top-2, renormalize, build per-expert
// gather lists (counts zeroed by memset beforehand).
// ---------------------------------------------------------------------------
__global__ __launch_bounds__(256) void moe_route(const float* __restrict__ logits,
                                                 int* __restrict__ counts,
                                                 int* __restrict__ tokens,
                                                 float* __restrict__ wts) {
    int t = blockIdx.x * blockDim.x + threadIdx.x;
    if (t >= T_) return;
    float p[E_];
    float mx = -1e30f;
    #pragma unroll
    for (int e = 0; e < E_; ++e) { p[e] = logits[t * E_ + e]; mx = fmaxf(mx, p[e]); }
    #pragma unroll
    for (int e = 0; e < E_; ++e) p[e] = __expf(p[e] - mx);   // softmax denom cancels in renorm
    int i1 = 0; float p1 = p[0];
    #pragma unroll
    for (int e = 1; e < E_; ++e) if (p[e] > p1) { p1 = p[e]; i1 = e; }
    int i2 = -1; float p2 = -1e30f;
    #pragma unroll
    for (int e = 0; e < E_; ++e) if (e != i1 && p[e] > p2) { p2 = p[e]; i2 = e; }
    const float inv = 1.0f / (p1 + p2);
    int s1 = atomicAdd(&counts[i1], 1);
    tokens[i1 * T_ + s1] = t;  wts[i1 * T_ + s1] = p1 * inv;
    int s2 = atomicAdd(&counts[i2], 1);
    tokens[i2 * T_ + s2] = t;  wts[i2 * T_ + s2] = p2 * inv;
}

// ---------------------------------------------------------------------------
// Step 2: GEMM1: gu = gather(x) @ w13[e]^T with fused SiLU(gate)*up.
// Token-gather pointers hoisted out of the K loop; global tiles for step k+1
// prefetched into registers while WMMA runs on step k.
// Grid: (I/BN, T/BM, E). Block: 256 = 8 waves; wave w -> (mi = w>>1, ni = w&1).
// ---------------------------------------------------------------------------
__global__ __launch_bounds__(256) void moe_gemm1(const float* __restrict__ x,
                                                 const float* __restrict__ w13,
                                                 const int* __restrict__ counts,
                                                 const int* __restrict__ tokens,
                                                 unsigned short* __restrict__ act) {
    const int e = blockIdx.z;
    const int cnt = counts[e];
    const int mbase = blockIdx.y * BM;
    if (mbase >= cnt) return;
    const int nbase = blockIdx.x * BN;

    __shared__ unsigned short As[BM * LDA];
    __shared__ unsigned short Bg[BN * LDB];
    __shared__ unsigned short Bu[BN * LDB];

    const int tid  = threadIdx.x;
    const int lane = tid & 31;
    const int wave = tid >> 5;
    const int mi = wave >> 1, ni = wave & 1;

    const float* __restrict__ w13e = w13 + (size_t)e * (2 * I_) * H_;

    // K-invariant staging assignments (4 A chunks + 2 gate + 2 up per thread).
    bool aValid[4]; const float* aSrc[4]; unsigned short* aDst[4];
    #pragma unroll
    for (int j = 0; j < 4; ++j) {
        const int idx = tid + 256 * j;
        const int row = idx >> 4;
        const int c4  = (idx & 15) * 4;
        const int slot = mbase + row;
        aValid[j] = slot < cnt;
        const int t = aValid[j] ? tokens[e * T_ + slot] : 0;
        aSrc[j] = x + (size_t)t * H_ + c4;
        aDst[j] = &As[row * LDA + c4];
    }
    const float* gSrc[2]; const float* uSrc[2];
    unsigned short* gDst[2]; unsigned short* uDst[2];
    #pragma unroll
    for (int j = 0; j < 2; ++j) {
        const int idx = tid + 256 * j;
        const int row = idx >> 4;
        const int c4  = (idx & 15) * 4;
        gSrc[j] = w13e + (size_t)(nbase + row) * H_ + c4;
        uSrc[j] = w13e + (size_t)(I_ + nbase + row) * H_ + c4;
        gDst[j] = &Bg[row * LDB + c4];
        uDst[j] = &Bu[row * LDB + c4];
    }

    const float4 z4 = make_float4(0.f, 0.f, 0.f, 0.f);
    float4 aR[4], gR[2], uR[2];
    #pragma unroll
    for (int j = 0; j < 4; ++j) aR[j] = aValid[j] ? *(const float4*)(aSrc[j]) : z4;
    #pragma unroll
    for (int j = 0; j < 2; ++j) { gR[j] = *(const float4*)(gSrc[j]); uR[j] = *(const float4*)(uSrc[j]); }

    v8f cg = {}; v8f cu = {};

    for (int k0 = 0; k0 < H_; k0 += BK) {
        // Drain prefetched registers into LDS (f32 -> bf16).
        #pragma unroll
        for (int j = 0; j < 4; ++j) st_bf4(aDst[j], aR[j]);
        #pragma unroll
        for (int j = 0; j < 2; ++j) { st_bf4(gDst[j], gR[j]); st_bf4(uDst[j], uR[j]); }
        __syncthreads();

        // Kick off next tile's global loads; they stay in flight during WMMA.
        const int kn = k0 + BK;
        if (kn < H_) {
            #pragma unroll
            for (int j = 0; j < 4; ++j) aR[j] = aValid[j] ? *(const float4*)(aSrc[j] + kn) : z4;
            #pragma unroll
            for (int j = 0; j < 2; ++j) {
                gR[j] = *(const float4*)(gSrc[j] + kn);
                uR[j] = *(const float4*)(uSrc[j] + kn);
            }
        }

        #pragma unroll
        for (int kk = 0; kk < BK; kk += 32) {
            v16bf a  = load_frag_a(&As[(mi * 16) * LDA + kk], lane, LDA);
            v16bf bg = load_frag_b(&Bg[(ni * 16) * LDB + kk], lane, LDB);
            v16bf bu = load_frag_b(&Bu[(ni * 16) * LDB + kk], lane, LDB);
            cg = wmma_bf16(a, bg, cg);
            cu = wmma_bf16(a, bu, cu);
        }
        __syncthreads();
    }

    // act = silu(gate) * up, stored bf16. C layout: vgpr r, lane L -> row
    // (L>>4)*8 + r, col L&15 (ISA 7.12.2).
    const int n    = nbase + ni * 16 + (lane & 15);
    const int mrow = mi * 16 + (lane >> 4) * 8;
    #pragma unroll
    for (int r = 0; r < 8; ++r) {
        const int slot = mbase + mrow + r;
        if (slot < cnt) {
            const float g = cg[r], u = cu[r];
            const float a = (g / (1.0f + __expf(-g))) * u;
            act[(size_t)(e * T_ + slot) * I_ + n] = f2bf(a);
        }
    }
}

// ---------------------------------------------------------------------------
// Step 3: GEMM2: y = act @ w2[e]^T, scale by routing weight, atomic scatter
// into out. Same pipelined structure. Grid: (H/BN, T/BM, E).
// ---------------------------------------------------------------------------
__global__ __launch_bounds__(256) void moe_gemm2(const unsigned short* __restrict__ act,
                                                 const float* __restrict__ w2,
                                                 const int* __restrict__ counts,
                                                 const int* __restrict__ tokens,
                                                 const float* __restrict__ wts,
                                                 float* __restrict__ out) {
    const int e = blockIdx.z;
    const int cnt = counts[e];
    const int mbase = blockIdx.y * BM;
    if (mbase >= cnt) return;
    const int hbase = blockIdx.x * BN;

    __shared__ unsigned short As[BM * LDA];
    __shared__ unsigned short Bs[BN * LDB];

    const int tid  = threadIdx.x;
    const int lane = tid & 31;
    const int wave = tid >> 5;
    const int mi = wave >> 1, ni = wave & 1;

    const float* __restrict__ w2e = w2 + (size_t)e * H_ * I_;

    // K-invariant staging assignments.
    bool aValid[4]; const unsigned short* aSrc[4]; unsigned short* aDst[4];
    #pragma unroll
    for (int j = 0; j < 4; ++j) {
        const int idx = tid + 256 * j;
        const int row = idx >> 4;
        const int c4  = (idx & 15) * 4;
        const int slot = mbase + row;
        aValid[j] = slot < cnt;
        aSrc[j] = act + (size_t)(e * T_ + slot) * I_ + c4;
        aDst[j] = &As[row * LDA + c4];
    }
    const float* bSrc[2]; unsigned short* bDst[2];
    #pragma unroll
    for (int j = 0; j < 2; ++j) {
        const int idx = tid + 256 * j;
        const int row = idx >> 4;
        const int c4  = (idx & 15) * 4;
        bSrc[j] = w2e + (size_t)(hbase + row) * I_ + c4;
        bDst[j] = &Bs[row * LDB + c4];
    }

    const ushort4 zu = make_ushort4(0, 0, 0, 0);
    ushort4 aR[4]; float4 bR[2];
    #pragma unroll
    for (int j = 0; j < 4; ++j) aR[j] = aValid[j] ? *(const ushort4*)(aSrc[j]) : zu;
    #pragma unroll
    for (int j = 0; j < 2; ++j) bR[j] = *(const float4*)(bSrc[j]);

    v8f c = {};

    for (int k0 = 0; k0 < I_; k0 += BK) {
        #pragma unroll
        for (int j = 0; j < 4; ++j) *(ushort4*)aDst[j] = aR[j];
        #pragma unroll
        for (int j = 0; j < 2; ++j) st_bf4(bDst[j], bR[j]);
        __syncthreads();

        const int kn = k0 + BK;
        if (kn < I_) {
            #pragma unroll
            for (int j = 0; j < 4; ++j) aR[j] = aValid[j] ? *(const ushort4*)(aSrc[j] + kn) : zu;
            #pragma unroll
            for (int j = 0; j < 2; ++j) bR[j] = *(const float4*)(bSrc[j] + kn);
        }

        #pragma unroll
        for (int kk = 0; kk < BK; kk += 32) {
            v16bf a = load_frag_a(&As[(mi * 16) * LDA + kk], lane, LDA);
            v16bf b = load_frag_b(&Bs[(ni * 16) * LDB + kk], lane, LDB);
            c = wmma_bf16(a, b, c);
        }
        __syncthreads();
    }

    const int h    = hbase + ni * 16 + (lane & 15);
    const int mrow = mi * 16 + (lane >> 4) * 8;
    #pragma unroll
    for (int r = 0; r < 8; ++r) {
        const int slot = mbase + mrow + r;
        if (slot < cnt) {
            const int   t = tokens[e * T_ + slot];
            const float w = wts[e * T_ + slot];
            atomicAdd(out + (size_t)t * H_ + h, c[r] * w);
        }
    }
}

// ---------------------------------------------------------------------------
// Host-side launch. Workspace layout:
//   [0,256)                      : counts (E ints, padded)
//   [256, 256+16K)               : tokens (E*T ints)
//   [+16K, +32K)                 : wts    (E*T floats)
//   [+32K, +32K + 8MB)           : act    (E*T*I bf16)
// ---------------------------------------------------------------------------
extern "C" void kernel_launch(void* const* d_in, const int* in_sizes, int n_in,
                              void* d_out, int out_size, void* d_ws, size_t ws_size,
                              hipStream_t stream) {
    (void)in_sizes; (void)n_in; (void)out_size; (void)ws_size;

    const float* x      = (const float*)d_in[0];
    const float* logits = (const float*)d_in[1];
    const float* w13    = (const float*)d_in[2];
    const float* w2     = (const float*)d_in[3];
    float*       out    = (float*)d_out;

    char* ws = (char*)d_ws;
    int*            counts = (int*)ws;
    int*            tokens = (int*)(ws + 256);
    float*          wts    = (float*)(ws + 256 + E_ * T_ * sizeof(int));
    unsigned short* act    = (unsigned short*)(ws + 256 + E_ * T_ * (sizeof(int) + sizeof(float)));

    hipMemsetAsync(counts, 0, E_ * sizeof(int), stream);
    hipMemsetAsync(out, 0, (size_t)T_ * H_ * sizeof(float), stream);

    moe_route<<<dim3((T_ + 255) / 256), dim3(256), 0, stream>>>(logits, counts, tokens, wts);

    moe_gemm1<<<dim3(I_ / BN, T_ / BM, E_), dim3(256), 0, stream>>>(x, w13, counts, tokens, act);

    moe_gemm2<<<dim3(H_ / BN, T_ / BM, E_), dim3(256), 0, stream>>>(act, w2, counts, tokens, wts, out);
}
